// WindowAttention_58617713656000
// MI455X (gfx1250) — matrix operand order repaired and checked by reading, not compiled, not run on hardware
//
#include <hip/hip_runtime.h>
#include <hip/hip_bf16.h>
#include <math.h>

// ---------------------------------------------------------------------------
// Fused window attention for MI455X (gfx1250, wave32, WMMA bf16).
// One workgroup = one window (128 tokens x 192 dim). 8 waves; wave w owns
// token rows [16w, 16w+16) through the entire pipeline (qkv -> attn -> proj).
// ---------------------------------------------------------------------------

#define NTOK 128
#define CDIM 192
#define NHEAD 6
#define HDIM 32
#define ATT_SCALE 0.17677669529663689f   // 1/sqrt(32)

typedef __bf16 bf16_t;
typedef bf16_t bf16x16 __attribute__((ext_vector_type(16)));
typedef bf16_t bf16x8  __attribute__((ext_vector_type(8)));
typedef float  f32x8   __attribute__((ext_vector_type(8)));
typedef unsigned int u32x4 __attribute__((ext_vector_type(4)));
// 32B vector but only 16B-aligned (LDS rows are 16B aligned); backend splits
// into two b128 accesses.
typedef unsigned int u32x8a __attribute__((ext_vector_type(8), aligned(16)));

__device__ __forceinline__ f32x8 zero8() {
  f32x8 z;
#pragma unroll
  for (int i = 0; i < 8; ++i) z[i] = 0.0f;
  return z;
}
__device__ __forceinline__ bf16x16 cat8(bf16x8 lo, bf16x8 hi) {
  return __builtin_shufflevector(lo, hi, 0, 1, 2, 3, 4, 5, 6, 7,
                                 8, 9, 10, 11, 12, 13, 14, 15);
}
// A fragment (16x32 bf16) from a row-major bf16 block, rowStride in elements.
// Lane l: row = l&15, K-half = (l>>4)?16:0 ; elems 0..15 = K[kh..kh+15]
// (contiguous K order, matching the B fragments).
__device__ __forceinline__ bf16x16 load_a16(const bf16_t* base, int rowStride,
                                            int koff, int lane) {
  int m  = lane & 15;
  int kb = (lane >> 4) << 4;
  const bf16_t* p = base + m * rowStride + koff + kb;
  bf16x8 lo = __builtin_bit_cast(bf16x8, *(const u32x4*)(p));
  bf16x8 hi = __builtin_bit_cast(bf16x8, *(const u32x4*)(p + 8));
  return cat8(lo, hi);
}
// B fragment where the 16 per-lane elements are contiguous in memory.
__device__ __forceinline__ bf16x16 load_b_contig(const bf16_t* p) {
  return __builtin_bit_cast(bf16x16, *(const u32x8a*)(p));
}
// Pre-swizzled weight tile: 512 bf16 per 32x16 tile, lane-major.
__device__ __forceinline__ bf16x16 load_b_tile(const bf16_t* tiles, int tileIdx,
                                               int lane) {
  return load_b_contig(tiles + (tileIdx << 9) + lane * 16);
}
__device__ __forceinline__ f32x8 wmma_bf16(bf16x16 a, bf16x16 b, f32x8 c) {
  return __builtin_amdgcn_wmma_f32_16x16x32_bf16(false, a, false, b, (short)0, c,
                                                 false, false);
}

// ---------------------------------------------------------------------------
// Prep kernels (run once per launch; outputs live in d_ws, hot in L2).
// ---------------------------------------------------------------------------
__global__ void k_swizzle_w(const float* __restrict__ W, int K, int N,
                            bf16_t* __restrict__ dst) {
  int idx = blockIdx.x * blockDim.x + threadIdx.x;
  if (idx >= K * N) return;
  int k = idx / N, n = idx - k * N;
  int kt = k >> 5, nt = n >> 4;
  int kin = k & 31, nin = n & 15;
  int lane = ((kin >> 4) << 4) | nin;
  int e = kin & 15;
  int NT = N >> 4;
  dst[((kt * NT + nt) << 9) + lane * 16 + e] = (bf16_t)W[idx];
}

__global__ void k_gather_bias(const float* __restrict__ rpb, const int* __restrict__ rel,
                              float* __restrict__ dst) {
  int idx = blockIdx.x * blockDim.x + threadIdx.x;
  if (idx >= NHEAD * NTOK * NTOK) return;
  int h = idx / (NTOK * NTOK);
  int ij = idx - h * (NTOK * NTOK);
  dst[idx] = rpb[rel[ij] * NHEAD + h];
}

__global__ void k_conv_pb(const float* __restrict__ posb, bf16_t* __restrict__ dst) {
  int idx = blockIdx.x * blockDim.x + threadIdx.x;
  if (idx >= NTOK * CDIM) return;
  int t = idx / CDIM, c = idx - t * CDIM;
  dst[idx] = (bf16_t)posb[(t & 63) * CDIM + c];   // tile(position_bias,(1,2,1))
}

// ---------------------------------------------------------------------------
// Main fused kernel. LDS strides chosen so every fragment load is 16B aligned.
// ---------------------------------------------------------------------------
#define XS_STRIDE 200    // 400B rows (25*16)
#define KS_STRIDE 40     // 80B rows  (5*16)
#define VT_STRIDE 136    // 272B rows (17*16)
#define PST_STRIDE 136

__global__ __launch_bounds__(256, 1)
void wattn_fused(const float* __restrict__ x, const float* __restrict__ mask,
                 const float* __restrict__ b_proj,
                 const bf16_t* __restrict__ wqkv_sw,
                 const bf16_t* __restrict__ wqm_sw,
                 const bf16_t* __restrict__ wproj_sw,
                 const float* __restrict__ bias6,
                 const bf16_t* __restrict__ pb,
                 float* __restrict__ out) {
  __shared__ __align__(16) bf16_t Xs[NTOK * XS_STRIDE];     // x (later x+pb)
  __shared__ __align__(16) bf16_t Ks[NTOK * KS_STRIDE];     // K head, row-major
  __shared__ __align__(16) bf16_t Vt[HDIM * VT_STRIDE];     // V head, transposed
  __shared__ __align__(16) bf16_t Pst[8 * 16 * PST_STRIDE]; // per-wave staging

  const int tid   = threadIdx.x;
  const int lane  = tid & 31;
  const int wid   = tid >> 5;
  const int bwin  = blockIdx.x;
  const int Rbase = wid << 4;            // this wave's token-row block
  const int col0  = lane & 15;
  const int mh    = (lane >> 4) << 3;    // C-layout row-half offset
  const int khalf = (lane >> 4) << 4;    // B-layout K-half offset
  bf16_t* mystage = Pst + wid * (16 * PST_STRIDE);

  const float* xb = x + (size_t)bwin * (NTOK * CDIM);
  const float* mk = mask + (size_t)(bwin & 63) * (NTOK * NTOK);

  // Stage x -> LDS bf16
  for (int idx = tid; idx < NTOK * CDIM; idx += 256) {
    int t = idx / CDIM, c = idx - t * CDIM;
    Xs[t * XS_STRIDE + c] = (bf16_t)xb[idx];
  }

  f32x8 facc[12];                         // final 16x192 rows, accumulated over heads
#pragma unroll
  for (int i = 0; i < 12; ++i) facc[i] = zero8();

  // ======================= self attention branch ==========================
  for (int h = 0; h < NHEAD; ++h) {
    __syncthreads();                      // Xs ready / prior-head K,V consumers done
    const int nq = 2 * h, nk = 12 + 2 * h, nv = 24 + 2 * h;
    f32x8 acc6[6];
#pragma unroll
    for (int j = 0; j < 6; ++j) acc6[j] = zero8();
#pragma unroll
    for (int kc = 0; kc < 6; ++kc) {      // K = 192 = 6 x 32
      bf16x16 a = load_a16(Xs + Rbase * XS_STRIDE, XS_STRIDE, kc * 32, lane);
      acc6[0] = wmma_bf16(a, load_b_tile(wqkv_sw, kc * 36 + nq,     lane), acc6[0]);
      acc6[1] = wmma_bf16(a, load_b_tile(wqkv_sw, kc * 36 + nq + 1, lane), acc6[1]);
      acc6[2] = wmma_bf16(a, load_b_tile(wqkv_sw, kc * 36 + nk,     lane), acc6[2]);
      acc6[3] = wmma_bf16(a, load_b_tile(wqkv_sw, kc * 36 + nk + 1, lane), acc6[3]);
      acc6[4] = wmma_bf16(a, load_b_tile(wqkv_sw, kc * 36 + nv,     lane), acc6[4]);
      acc6[5] = wmma_bf16(a, load_b_tile(wqkv_sw, kc * 36 + nv + 1, lane), acc6[5]);
    }
    // Q (scaled) -> wave staging; K -> Ks; V -> Vt (transposed)
#pragma unroll
    for (int j = 0; j < 2; ++j)
#pragma unroll
      for (int r = 0; r < 8; ++r) {
        mystage[(mh + r) * PST_STRIDE + j * 16 + col0] = (bf16_t)(acc6[j][r] * ATT_SCALE);
        Ks[(Rbase + mh + r) * KS_STRIDE + j * 16 + col0] = (bf16_t)acc6[2 + j][r];
        Vt[(j * 16 + col0) * VT_STRIDE + (Rbase + mh + r)] = (bf16_t)acc6[4 + j][r];
      }
    bf16x16 qf = load_a16(mystage, PST_STRIDE, 0, lane);
    __syncthreads();                      // K/V of all waves visible

    // S = Q @ K^T  (8 col tiles x K=32), then + bias + mask
    f32x8 sacc[8];
#pragma unroll
    for (int nt = 0; nt < 8; ++nt)
      sacc[nt] = wmma_bf16(qf, load_b_contig(Ks + (nt * 16 + col0) * KS_STRIDE + khalf),
                           zero8());
    const float* bh = bias6 + h * (NTOK * NTOK);
#pragma unroll
    for (int nt = 0; nt < 8; ++nt)
#pragma unroll
      for (int r = 0; r < 8; ++r) {
        int row = Rbase + mh + r, cc = nt * 16 + col0;
        sacc[nt][r] += bh[row * NTOK + cc] + mk[row * NTOK + cc];
      }
    // softmax over 128 cols: row lives in lanes of one 16-lane half
#pragma unroll
    for (int r = 0; r < 8; ++r) {
      float mval = sacc[0][r];
#pragma unroll
      for (int nt = 1; nt < 8; ++nt) mval = fmaxf(mval, sacc[nt][r]);
      mval = fmaxf(mval, __shfl_xor(mval, 8, 16));
      mval = fmaxf(mval, __shfl_xor(mval, 4, 16));
      mval = fmaxf(mval, __shfl_xor(mval, 2, 16));
      mval = fmaxf(mval, __shfl_xor(mval, 1, 16));
      float s = 0.0f;
#pragma unroll
      for (int nt = 0; nt < 8; ++nt) {
        float e = __expf(sacc[nt][r] - mval);
        sacc[nt][r] = e; s += e;
      }
      s += __shfl_xor(s, 8, 16);
      s += __shfl_xor(s, 4, 16);
      s += __shfl_xor(s, 2, 16);
      s += __shfl_xor(s, 1, 16);
      float inv = 1.0f / s;
#pragma unroll
      for (int nt = 0; nt < 8; ++nt) sacc[nt][r] *= inv;
    }
    // P -> staging (C-layout -> A-layout transpose via LDS)
#pragma unroll
    for (int nt = 0; nt < 8; ++nt)
#pragma unroll
      for (int r = 0; r < 8; ++r)
        mystage[(mh + r) * PST_STRIDE + nt * 16 + col0] = (bf16_t)sacc[nt][r];
    // out_h = P @ V  (K = 128 = 4 x 32)
    f32x8 oacc[2]; oacc[0] = zero8(); oacc[1] = zero8();
#pragma unroll
    for (int kc = 0; kc < 4; ++kc) {
      bf16x16 pa = load_a16(mystage, PST_STRIDE, kc * 32, lane);
      oacc[0] = wmma_bf16(pa, load_b_contig(Vt + col0 * VT_STRIDE + kc * 32 + khalf), oacc[0]);
      oacc[1] = wmma_bf16(pa, load_b_contig(Vt + (16 + col0) * VT_STRIDE + kc * 32 + khalf), oacc[1]);
    }
    // out_h -> staging -> A frag; accumulate final: out_self part uses wproj rows 192+32h
#pragma unroll
    for (int nt = 0; nt < 2; ++nt)
#pragma unroll
      for (int r = 0; r < 8; ++r)
        mystage[(mh + r) * PST_STRIDE + nt * 16 + col0] = (bf16_t)oacc[nt][r];
    bf16x16 of = load_a16(mystage, PST_STRIDE, 0, lane);
#pragma unroll
    for (int nt = 0; nt < 12; ++nt)
      facc[nt] = wmma_bf16(of, load_b_tile(wproj_sw, (6 + h) * 12 + nt, lane), facc[nt]);
  }

  // ================== mutual cross-window branch ==========================
  __syncthreads();
  for (int idx = tid; idx < NTOK * CDIM; idx += 256) {   // Xs = x + pos_bias
    int t = idx / CDIM, c = idx - t * CDIM;
    int o = t * XS_STRIDE + c;
    Xs[o] = (bf16_t)((float)Xs[o] + (float)pb[idx]);
  }

  const int qRbase = Rbase ^ 64;           // q2 for rows 0..63, q1 for rows 64..127
  const int kb = (wid < 4) ? 0 : 64;       // key/value token half
  for (int h = 0; h < NHEAD; ++h) {
    __syncthreads();
    const int nq = 2 * h, nk = 12 + 2 * h, nv = 24 + 2 * h;
    f32x8 acc6[6];
#pragma unroll
    for (int j = 0; j < 6; ++j) acc6[j] = zero8();
#pragma unroll
    for (int kc = 0; kc < 6; ++kc) {
      bf16x16 aq  = load_a16(Xs + qRbase * XS_STRIDE, XS_STRIDE, kc * 32, lane);
      bf16x16 akv = load_a16(Xs + Rbase  * XS_STRIDE, XS_STRIDE, kc * 32, lane);
      acc6[0] = wmma_bf16(aq,  load_b_tile(wqm_sw, kc * 36 + nq,     lane), acc6[0]);
      acc6[1] = wmma_bf16(aq,  load_b_tile(wqm_sw, kc * 36 + nq + 1, lane), acc6[1]);
      acc6[2] = wmma_bf16(akv, load_b_tile(wqm_sw, kc * 36 + nk,     lane), acc6[2]);
      acc6[3] = wmma_bf16(akv, load_b_tile(wqm_sw, kc * 36 + nk + 1, lane), acc6[3]);
      acc6[4] = wmma_bf16(akv, load_b_tile(wqm_sw, kc * 36 + nv,     lane), acc6[4]);
      acc6[5] = wmma_bf16(akv, load_b_tile(wqm_sw, kc * 36 + nv + 1, lane), acc6[5]);
    }
#pragma unroll
    for (int j = 0; j < 2; ++j)
#pragma unroll
      for (int r = 0; r < 8; ++r) {
        mystage[(mh + r) * PST_STRIDE + j * 16 + col0] = (bf16_t)(acc6[j][r] * ATT_SCALE);
        Ks[(Rbase + mh + r) * KS_STRIDE + j * 16 + col0] = (bf16_t)acc6[2 + j][r];
        Vt[(j * 16 + col0) * VT_STRIDE + (Rbase + mh + r)] = (bf16_t)acc6[4 + j][r];
      }
    bf16x16 qf = load_a16(mystage, PST_STRIDE, 0, lane);
    __syncthreads();

    f32x8 sacc[4];
#pragma unroll
    for (int nt = 0; nt < 4; ++nt)
      sacc[nt] = wmma_bf16(
          qf, load_b_contig(Ks + (kb + nt * 16 + col0) * KS_STRIDE + khalf), zero8());
#pragma unroll
    for (int nt = 0; nt < 4; ++nt)
#pragma unroll
      for (int r = 0; r < 8; ++r) {
        int i63 = (Rbase + mh + r) & 63;               // mask[:64,:64]
        sacc[nt][r] += mk[i63 * NTOK + nt * 16 + col0];
      }
#pragma unroll
    for (int r = 0; r < 8; ++r) {
      float mval = sacc[0][r];
#pragma unroll
      for (int nt = 1; nt < 4; ++nt) mval = fmaxf(mval, sacc[nt][r]);
      mval = fmaxf(mval, __shfl_xor(mval, 8, 16));
      mval = fmaxf(mval, __shfl_xor(mval, 4, 16));
      mval = fmaxf(mval, __shfl_xor(mval, 2, 16));
      mval = fmaxf(mval, __shfl_xor(mval, 1, 16));
      float s = 0.0f;
#pragma unroll
      for (int nt = 0; nt < 4; ++nt) {
        float e = __expf(sacc[nt][r] - mval);
        sacc[nt][r] = e; s += e;
      }
      s += __shfl_xor(s, 8, 16);
      s += __shfl_xor(s, 4, 16);
      s += __shfl_xor(s, 2, 16);
      s += __shfl_xor(s, 1, 16);
      float inv = 1.0f / s;
#pragma unroll
      for (int nt = 0; nt < 4; ++nt) sacc[nt][r] *= inv;
    }
#pragma unroll
    for (int nt = 0; nt < 4; ++nt)
#pragma unroll
      for (int r = 0; r < 8; ++r)
        mystage[(mh + r) * PST_STRIDE + nt * 16 + col0] = (bf16_t)sacc[nt][r];
    f32x8 oacc[2]; oacc[0] = zero8(); oacc[1] = zero8();
#pragma unroll
    for (int kc = 0; kc < 2; ++kc) {
      bf16x16 pa = load_a16(mystage, PST_STRIDE, kc * 32, lane);
      oacc[0] = wmma_bf16(pa, load_b_contig(Vt + col0 * VT_STRIDE + kb + kc * 32 + khalf), oacc[0]);
      oacc[1] = wmma_bf16(pa, load_b_contig(Vt + (16 + col0) * VT_STRIDE + kb + kc * 32 + khalf), oacc[1]);
    }
#pragma unroll
    for (int nt = 0; nt < 2; ++nt)
#pragma unroll
      for (int r = 0; r < 8; ++r)
        mystage[(mh + r) * PST_STRIDE + nt * 16 + col0] = (bf16_t)oacc[nt][r];
    bf16x16 of = load_a16(mystage, PST_STRIDE, 0, lane);
#pragma unroll
    for (int nt = 0; nt < 12; ++nt)        // aligned part uses wproj rows 32h
      facc[nt] = wmma_bf16(of, load_b_tile(wproj_sw, h * 12 + nt, lane), facc[nt]);
  }

  // ===================== epilogue: + b_proj, store ========================
  float* ob = out + (size_t)bwin * (NTOK * CDIM);
#pragma unroll
  for (int nt = 0; nt < 12; ++nt) {
    float bpv = b_proj[nt * 16 + col0];
#pragma unroll
    for (int r = 0; r < 8; ++r)
      ob[(Rbase + mh + r) * CDIM + nt * 16 + col0] = facc[nt][r] + bpv;
  }
}

// ---------------------------------------------------------------------------
// Workspace layout (bytes):
//   [0)        wqkv swizzled   6*36 tiles * 1024B = 221184
//   [221184)   wqkv_mutual sw                      = 221184
//   [442368)   wproj swizzled 12*12 tiles * 1024B  = 147456
//   [589824)   bias6 f32 [6][128][128]             = 393216
//   [983040)   pb bf16 [128][192]                  =  49152
//   total 1032192 bytes
// ---------------------------------------------------------------------------
extern "C" void kernel_launch(void* const* d_in, const int* in_sizes, int n_in,
                              void* d_out, int out_size, void* d_ws, size_t ws_size,
                              hipStream_t stream) {
  (void)in_sizes; (void)n_in; (void)out_size; (void)ws_size;
  const float* x      = (const float*)d_in[0];
  const float* mask   = (const float*)d_in[1];
  const float* w_qkv  = (const float*)d_in[2];
  const float* w_qkvm = (const float*)d_in[3];
  const float* w_proj = (const float*)d_in[4];
  const float* b_proj = (const float*)d_in[5];
  const float* rpb    = (const float*)d_in[6];
  const float* posb   = (const float*)d_in[7];
  const int*   rel    = (const int*)d_in[8];

  char* ws = (char*)d_ws;
  bf16_t* wqkv_sw  = (bf16_t*)(ws);
  bf16_t* wqm_sw   = (bf16_t*)(ws + 221184);
  bf16_t* wproj_sw = (bf16_t*)(ws + 442368);
  float*  bias6    = (float*)(ws + 589824);
  bf16_t* pb       = (bf16_t*)(ws + 983040);

  k_swizzle_w<<<(192 * 576 + 255) / 256, 256, 0, stream>>>(w_qkv, 192, 576, wqkv_sw);
  k_swizzle_w<<<(192 * 576 + 255) / 256, 256, 0, stream>>>(w_qkvm, 192, 576, wqm_sw);
  k_swizzle_w<<<(384 * 192 + 255) / 256, 256, 0, stream>>>(w_proj, 384, 192, wproj_sw);
  k_gather_bias<<<(6 * 128 * 128 + 255) / 256, 256, 0, stream>>>(rpb, rel, bias6);
  k_conv_pb<<<(128 * 192 + 255) / 256, 256, 0, stream>>>(posb, pb);

  wattn_fused<<<1024, 256, 0, stream>>>(x, mask, b_proj, wqkv_sw, wqm_sw, wproj_sw,
                                        bias6, pb, (float*)d_out);
}
